// GRU_6081673691645
// MI455X (gfx1250) — matrix operand order repaired and checked by reading, not compiled, not run on hardware
//
#include <hip/hip_runtime.h>
#include <math.h>

#define T_SEQ 64
#define BATCH 128
#define VOCAB 10000
#define HID   1024
#define NLAY  2

typedef __attribute__((ext_vector_type(16))) __bf16 v16bf;
typedef __attribute__((ext_vector_type(8)))  __bf16 v8bf;
typedef __attribute__((ext_vector_type(8)))  float  v8f;

__device__ __forceinline__ v8f wmma_bf16(v16bf a, v16bf b, v8f c) {
  return __builtin_amdgcn_wmma_f32_16x16x32_bf16(false, a, false, b, (short)0, c,
                                                 false, false);
}

// CDNA5 async global->LDS copy (VGLOBAL encoding, ASYNCcnt tracked).
// Operand order per ISA: VDST = LDS byte address VGPR, then global VADDR.
__device__ __forceinline__ void async_copy_b128(unsigned lds_off,
                                                const void* gptr) {
  asm volatile("global_load_async_to_lds_b128 %0, %1, off"
               :
               : "v"(lds_off), "v"(gptr)
               : "memory");
}
__device__ __forceinline__ void wait_async_all() {
  asm volatile("s_wait_asynccnt 0" ::: "memory");
}

// A fragment from a row-major LDS tile (stride 256 bf16 per row).
// Lane L needs rows m=L&15, k runs {half*16..+15}:
// two contiguous 16B chunks -> 2x ds_load_b128, no packing VALU.
__device__ __forceinline__ v16bf frag_a(const __bf16* Alds, int kk, int lane) {
  const __bf16* p = Alds + (lane & 15) * 256 + kk + ((lane >> 4) << 4);
  v8bf lo = *(const v8bf*)p;
  v8bf hi = *(const v8bf*)(p + 8);
  return __builtin_shufflevector(lo, hi, 0, 1, 2, 3, 4, 5, 6, 7, 8, 9, 10, 11,
                                 12, 13, 14, 15);
}

// B fragment from fragment-swizzled weights: block (nt,ks) = 32 lanes x 16 bf16,
// lane L holds k=(L>>4)*16+e, n=nt*16+(L&15). One 32B load -> 2x global_load_b128.
__device__ __forceinline__ v16bf frag_b(const __bf16* Wsw, int nt, int ks,
                                        int KS, int lane) {
  return *(const v16bf*)(Wsw + ((size_t)(nt * KS + ks) << 9) + lane * 16);
}

// ---------------------------------------------------------------------------
// One-time conversion / swizzle kernels
// ---------------------------------------------------------------------------
__global__ void swizzle_weight_kernel(const float* __restrict__ W,
                                      __bf16* __restrict__ Wsw, int K, int N,
                                      int total_groups) {
  const int g = blockIdx.x * blockDim.x + threadIdx.x;
  if (g >= total_groups) return;
  const int lane = g & 31;
  const int KS = K >> 5;
  const int t = g >> 5;
  const int ks = t % KS, nt = t / KS;
  int n = nt * 16 + (lane & 15);
  if (n >= N) n = N - 1;  // pad tiles read duplicated columns (stores guarded)
  const int kb = ks * 32 + ((lane >> 4) << 4);
  __bf16* d = Wsw + (size_t)g * 16;
#pragma unroll
  for (int e = 0; e < 16; ++e) d[e] = (__bf16)W[(size_t)(kb + e) * N + n];
}

__global__ void embed_kernel(const int* __restrict__ ids,
                             const float* __restrict__ emb,
                             __bf16* __restrict__ x) {
  const int tb = blockIdx.x;  // 0 .. T*B-1
  const float4 v = ((const float4*)(emb + (size_t)ids[tb] * HID))[threadIdx.x];
  __bf16 o[4] = {(__bf16)v.x, (__bf16)v.y, (__bf16)v.z, (__bf16)v.w};
  *(uint2*)(x + (size_t)tb * HID + threadIdx.x * 4) = *(uint2*)o;
}

__global__ void init_h_kernel(const float* __restrict__ h0,
                              float* __restrict__ hf, __bf16* __restrict__ hb,
                              int n) {
  int i = blockIdx.x * blockDim.x + threadIdx.x;
  const int stride = gridDim.x * blockDim.x;
  for (; i < n; i += stride) { float v = h0[i]; hf[i] = v; hb[i] = (__bf16)v; }
}

__global__ void copy_f32_kernel(const float* __restrict__ s,
                                float* __restrict__ d, int n) {
  int i = blockIdx.x * blockDim.x + threadIdx.x;
  const int stride = gridDim.x * blockDim.x;
  for (; i < n; i += stride) d[i] = s[i];
}

// ---------------------------------------------------------------------------
// Fused r/z gates: r=tanh([inp|h]@Wr+br), z=tanh([inp|h]@Wz+bz).
// Block=128 (4 waves); each wave owns 2 N-tiles for BOTH matrices (4 WMMAs
// per A fragment). Double-buffered async A staging. Grid: (HID/128, BATCH/16).
// ---------------------------------------------------------------------------
__global__ __launch_bounds__(128) void gates_rz_kernel(
    const __bf16* __restrict__ inp, const __bf16* __restrict__ hprev,
    const float* __restrict__ hprev_f, const __bf16* __restrict__ Wr,
    const __bf16* __restrict__ Wz, const float* __restrict__ br,
    const float* __restrict__ bz, float* __restrict__ z_out,
    __bf16* __restrict__ rh_out) {
  __shared__ __align__(16) __bf16 Alds[2][16 * 256];
  const int tid = threadIdx.x, lane = tid & 31;
  const int wave = __builtin_amdgcn_readfirstlane(tid >> 5);
  const int m0 = blockIdx.y * 16;
  const int nt0 = blockIdx.x * 8 + wave * 2;
  const int KS = (2 * HID) >> 5;  // 64 k-steps
  const int NSUP = (2 * HID) / 256;

  auto src_for = [&](int s) -> const __bf16* {
    return (s < NSUP / 2) ? inp + (size_t)m0 * HID + s * 256
                          : hprev + (size_t)m0 * HID + (s - NSUP / 2) * 256;
  };
  auto stage = [&](int buf, const __bf16* src) {
    const unsigned base = (unsigned)(size_t)(const void*)&Alds[buf][0];
#pragma unroll
    for (int i = tid; i < 512; i += 128) {
      const int r = i >> 5, q = i & 31;
      async_copy_b128(base + (unsigned)i * 16, src + (size_t)r * HID + q * 8);
    }
  };

  v8f ar0 = {}, ar1 = {}, az0 = {}, az1 = {};
  stage(0, src_for(0));
  wait_async_all();
  __syncthreads();
  for (int s = 0; s < NSUP; ++s) {
    if (s + 1 < NSUP) {
      stage((s + 1) & 1, src_for(s + 1));
      const int ksn = ((s + 1) * 256) >> 5;  // prefetch next weight super-step
      __builtin_prefetch(Wr + ((size_t)(nt0 * KS + ksn) << 9) + lane * 16, 0, 1);
      __builtin_prefetch(Wz + ((size_t)(nt0 * KS + ksn) << 9) + lane * 16, 0, 1);
    }
    const __bf16* Ab = &Alds[s & 1][0];
#pragma unroll
    for (int kk = 0; kk < 256; kk += 32) {
      const int ks = (s * 256 + kk) >> 5;
      const v16bf a = frag_a(Ab, kk, lane);
      ar0 = wmma_bf16(a, frag_b(Wr, nt0, ks, KS, lane), ar0);
      ar1 = wmma_bf16(a, frag_b(Wr, nt0 + 1, ks, KS, lane), ar1);
      az0 = wmma_bf16(a, frag_b(Wz, nt0, ks, KS, lane), az0);
      az1 = wmma_bf16(a, frag_b(Wz, nt0 + 1, ks, KS, lane), az1);
    }
    wait_async_all();
    __syncthreads();
  }
  const int half = lane >> 4;
#pragma unroll
  for (int t = 0; t < 2; ++t) {
    const v8f& accr = t ? ar1 : ar0;
    const v8f& accz = t ? az1 : az0;
    const int nc = (nt0 + t) * 16 + (lane & 15);
    const float brv = br[nc], bzv = bz[nc];
#pragma unroll
    for (int v = 0; v < 8; ++v) {
      const int m = m0 + half * 8 + v;
      const size_t idx = (size_t)m * HID + nc;
      const float r = tanhf(accr[v] + brv);
      const float z = tanhf(accz[v] + bzv);
      z_out[idx] = z;
      rh_out[idx] = (__bf16)(r * hprev_f[idx]);
    }
  }
}

// ---------------------------------------------------------------------------
// g gate + state update: h~=tanh([inp|r*h]@Wg+bg); hn=(1-z)*h+z*h~.
// ---------------------------------------------------------------------------
__global__ __launch_bounds__(128) void gate_g_kernel(
    const __bf16* __restrict__ inp, const __bf16* __restrict__ rh,
    const float* hprev_f, const __bf16* __restrict__ Wg,
    const float* __restrict__ bg, const float* __restrict__ z_in,
    float* hnew_f, __bf16* __restrict__ hnew_bf, __bf16* seq_out) {
  __shared__ __align__(16) __bf16 Alds[2][16 * 256];
  const int tid = threadIdx.x, lane = tid & 31;
  const int wave = __builtin_amdgcn_readfirstlane(tid >> 5);
  const int m0 = blockIdx.y * 16;
  const int nt0 = blockIdx.x * 8 + wave * 2;
  const int KS = (2 * HID) >> 5;
  const int NSUP = (2 * HID) / 256;

  auto src_for = [&](int s) -> const __bf16* {
    return (s < NSUP / 2) ? inp + (size_t)m0 * HID + s * 256
                          : rh + (size_t)m0 * HID + (s - NSUP / 2) * 256;
  };
  auto stage = [&](int buf, const __bf16* src) {
    const unsigned base = (unsigned)(size_t)(const void*)&Alds[buf][0];
#pragma unroll
    for (int i = tid; i < 512; i += 128) {
      const int r = i >> 5, q = i & 31;
      async_copy_b128(base + (unsigned)i * 16, src + (size_t)r * HID + q * 8);
    }
  };

  v8f a0 = {}, a1 = {};
  stage(0, src_for(0));
  wait_async_all();
  __syncthreads();
  for (int s = 0; s < NSUP; ++s) {
    if (s + 1 < NSUP) {
      stage((s + 1) & 1, src_for(s + 1));
      __builtin_prefetch(
          Wg + ((size_t)(nt0 * KS + (((s + 1) * 256) >> 5)) << 9) + lane * 16, 0,
          1);
    }
    const __bf16* Ab = &Alds[s & 1][0];
#pragma unroll
    for (int kk = 0; kk < 256; kk += 32) {
      const int ks = (s * 256 + kk) >> 5;
      const v16bf a = frag_a(Ab, kk, lane);
      a0 = wmma_bf16(a, frag_b(Wg, nt0, ks, KS, lane), a0);
      a1 = wmma_bf16(a, frag_b(Wg, nt0 + 1, ks, KS, lane), a1);
    }
    wait_async_all();
    __syncthreads();
  }
  const int half = lane >> 4;
#pragma unroll
  for (int t = 0; t < 2; ++t) {
    const v8f& acc = t ? a1 : a0;
    const int nc = (nt0 + t) * 16 + (lane & 15);
    const float bgv = bg[nc];
#pragma unroll
    for (int v = 0; v < 8; ++v) {
      const int m = m0 + half * 8 + v;
      const size_t idx = (size_t)m * HID + nc;
      const float ht = tanhf(acc[v] + bgv);
      const float z = z_in[idx];
      const float hn = (1.0f - z) * hprev_f[idx] + z * ht;
      hnew_f[idx] = hn;
      hnew_bf[idx] = (__bf16)hn;
      if (seq_out) seq_out[idx] = (__bf16)hn;
    }
  }
}

// ---------------------------------------------------------------------------
// Output projection over all timesteps: logits = tanh(Hseq@Wout + bout).
// M=T*B=8192, N padded to 632 tiles, K=1024. Grid: (79, 512).
// ---------------------------------------------------------------------------
__global__ __launch_bounds__(128) void logits_kernel(
    const __bf16* __restrict__ seq, const __bf16* __restrict__ W,
    const float* __restrict__ bout, float* __restrict__ out) {
  __shared__ __align__(16) __bf16 Alds[2][16 * 256];
  const int tid = threadIdx.x, lane = tid & 31;
  const int wave = __builtin_amdgcn_readfirstlane(tid >> 5);
  const int m0 = blockIdx.y * 16;
  const int nt0 = blockIdx.x * 8 + wave * 2;
  const int KS = HID >> 5;  // 32 k-steps
  const int NSUP = HID / 256;

  auto stage = [&](int buf, int s) {
    const __bf16* src = seq + (size_t)m0 * HID + s * 256;
    const unsigned base = (unsigned)(size_t)(const void*)&Alds[buf][0];
#pragma unroll
    for (int i = tid; i < 512; i += 128) {
      const int r = i >> 5, q = i & 31;
      async_copy_b128(base + (unsigned)i * 16, src + (size_t)r * HID + q * 8);
    }
  };

  v8f a0 = {}, a1 = {};
  stage(0, 0);
  wait_async_all();
  __syncthreads();
  for (int s = 0; s < NSUP; ++s) {
    if (s + 1 < NSUP) {
      stage((s + 1) & 1, s + 1);
      __builtin_prefetch(
          W + ((size_t)(nt0 * KS + (((s + 1) * 256) >> 5)) << 9) + lane * 16, 0,
          1);
    }
    const __bf16* Ab = &Alds[s & 1][0];
#pragma unroll
    for (int kk = 0; kk < 256; kk += 32) {
      const int ks = (s * 256 + kk) >> 5;
      const v16bf a = frag_a(Ab, kk, lane);
      a0 = wmma_bf16(a, frag_b(W, nt0, ks, KS, lane), a0);
      a1 = wmma_bf16(a, frag_b(W, nt0 + 1, ks, KS, lane), a1);
    }
    wait_async_all();
    __syncthreads();
  }
  const int half = lane >> 4;
#pragma unroll
  for (int t = 0; t < 2; ++t) {
    const v8f& acc = t ? a1 : a0;
    const int nc = (nt0 + t) * 16 + (lane & 15);
    if (nc < VOCAB) {
      const float bv = bout[nc];
#pragma unroll
      for (int v = 0; v < 8; ++v)
        out[(size_t)(m0 + half * 8 + v) * VOCAB + nc] = tanhf(acc[v] + bv);
    }
  }
}

// ---------------------------------------------------------------------------
// Host-side orchestration
// ---------------------------------------------------------------------------
extern "C" void kernel_launch(void* const* d_in, const int* in_sizes, int n_in,
                              void* d_out, int out_size, void* d_ws,
                              size_t ws_size, hipStream_t stream) {
  const int*   inputs = (const int*)d_in[0];
  const float* hidden = (const float*)d_in[1];
  const float* emb    = (const float*)d_in[2];
  const float* Wr     = (const float*)d_in[3];
  const float* br     = (const float*)d_in[4];
  const float* Wz     = (const float*)d_in[5];
  const float* bz     = (const float*)d_in[6];
  const float* Wg     = (const float*)d_in[7];
  const float* bg     = (const float*)d_in[8];
  const float* Wout   = (const float*)d_in[9];
  const float* bout   = (const float*)d_in[10];
  float* out = (float*)d_out;

  char* ws = (char*)d_ws;
  size_t off = 0;
  auto alloc = [&](size_t bytes) -> void* {
    void* p = ws + off;
    off += (bytes + 255) & ~(size_t)255;
    return p;
  };
  const size_t GATE_W = (size_t)2 * HID * HID;          // per layer per matrix
  const int VTILES_PAD = 632;                           // 79 blocks * 8 tiles
  __bf16* Wr_sw   = (__bf16*)alloc(NLAY * GATE_W * 2);
  __bf16* Wz_sw   = (__bf16*)alloc(NLAY * GATE_W * 2);
  __bf16* Wg_sw   = (__bf16*)alloc(NLAY * GATE_W * 2);
  __bf16* Wout_sw = (__bf16*)alloc((size_t)VTILES_PAD * 32 * 512 * 2);
  __bf16* x_bf    = (__bf16*)alloc((size_t)T_SEQ * BATCH * HID * 2);
  __bf16* seq_bf  = (__bf16*)alloc((size_t)T_SEQ * BATCH * HID * 2);
  __bf16* h_bf    = (__bf16*)alloc((size_t)NLAY * BATCH * HID * 2);
  float*  h_f     = (float*) alloc((size_t)NLAY * BATCH * HID * 4);
  float*  z_f     = (float*) alloc((size_t)BATCH * HID * 4);
  __bf16* rh_bf   = (__bf16*)alloc((size_t)BATCH * HID * 2);
  (void)ws_size; (void)in_sizes; (void)n_in; (void)out_size;

  // 1) one-time fragment-swizzle of all weights to bf16 (amortized over T)
  const int gate_groups = (HID / 16) * ((2 * HID) / 32) * 32;  // 131072
  for (int l = 0; l < NLAY; ++l) {
    swizzle_weight_kernel<<<(gate_groups + 255) / 256, 256, 0, stream>>>(
        Wr + l * GATE_W, Wr_sw + l * GATE_W, 2 * HID, HID, gate_groups);
    swizzle_weight_kernel<<<(gate_groups + 255) / 256, 256, 0, stream>>>(
        Wz + l * GATE_W, Wz_sw + l * GATE_W, 2 * HID, HID, gate_groups);
    swizzle_weight_kernel<<<(gate_groups + 255) / 256, 256, 0, stream>>>(
        Wg + l * GATE_W, Wg_sw + l * GATE_W, 2 * HID, HID, gate_groups);
  }
  const int vout_groups = VTILES_PAD * (HID / 32) * 32;  // 647168
  swizzle_weight_kernel<<<(vout_groups + 255) / 256, 256, 0, stream>>>(
      Wout, Wout_sw, HID, VOCAB, vout_groups);
  init_h_kernel<<<1024, 256, 0, stream>>>(hidden, h_f, h_bf, NLAY * BATCH * HID);

  // 2) embedding gather -> bf16 activations
  embed_kernel<<<T_SEQ * BATCH, 256, 0, stream>>>(inputs, emb, x_bf);

  // 3) recurrent loop (sequential dependency on h)
  const dim3 ggrid(HID / 128, BATCH / 16);
  for (int t = 0; t < T_SEQ; ++t) {
    const __bf16* inp = x_bf + (size_t)t * BATCH * HID;
    for (int l = 0; l < NLAY; ++l) {
      const size_t wo = (size_t)l * GATE_W;
      const size_t bo = (size_t)l * HID;
      const size_t ho = (size_t)l * BATCH * HID;
      __bf16* seq = (l == NLAY - 1) ? (seq_bf + (size_t)t * BATCH * HID) : nullptr;
      gates_rz_kernel<<<ggrid, 128, 0, stream>>>(
          inp, h_bf + ho, h_f + ho, Wr_sw + wo, Wz_sw + wo, br + bo, bz + bo,
          z_f, rh_bf);
      gate_g_kernel<<<ggrid, 128, 0, stream>>>(
          inp, rh_bf, h_f + ho, Wg_sw + wo, bg + bo, z_f, h_f + ho, h_bf + ho,
          seq);
      inp = h_bf + ho;
    }
  }

  // 4) one big projection GEMM + tanh over all timesteps
  logits_kernel<<<dim3(VTILES_PAD / 8, (T_SEQ * BATCH) / 16), 128, 0, stream>>>(
      seq_bf, Wout_sw, bout, out);

  // 5) final hidden state appended after logits
  copy_f32_kernel<<<1024, 256, 0, stream>>>(
      h_f, out + (size_t)T_SEQ * BATCH * VOCAB, NLAY * BATCH * HID);
}
